// UserAwareGate_12635793784885
// MI455X (gfx1250) — matrix operand order repaired and checked by reading, not compiled, not run on hardware
//
#include <hip/hip_runtime.h>
#include <hip/hip_bf16.h>

typedef __attribute__((ext_vector_type(16))) __bf16 v16bf;
typedef __attribute__((ext_vector_type(2)))  __bf16 v2bf;
typedef __attribute__((ext_vector_type(8)))  float  v8f;

#define EMB_DIM      2048
#define USER_DIM     256
#define TOT_K        (EMB_DIM + USER_DIM)          // 2304
#define NUM_EXPERTS  64
#define TOPK         8
#define KCHUNK       1152                          // K columns staged per LDS chunk
#define KB_PER_CHUNK (KCHUNK / 16)                 // 72
#define CHUNK_SLOTS  (KB_PER_CHUNK * NUM_EXPERTS)  // 4608
#define LDS_BYTES    (CHUNK_SLOTS * 32)            // 147456 B -> 2 workgroups/WGP

// f32 pair -> packed bf16 dword; plain casts let the backend pick native cvt ops
static __device__ __forceinline__ unsigned pk_bf16(float lo, float hi) {
  v2bf p;
  p[0] = (__bf16)lo;
  p[1] = (__bf16)hi;
  return __builtin_bit_cast(unsigned, p);
}

__global__ __launch_bounds__(256)
void UserAwareGate_kernel(const float* __restrict__ H, const float* __restrict__ U,
                          const float* __restrict__ W, const float* __restrict__ B,
                          float* __restrict__ out, int ntok) {
  extern __shared__ unsigned char wl[];

  const int lane = threadIdx.x & 31;
  const int wave = threadIdx.x >> 5;
  const int half = lane >> 4;        // which 16-lane half of the wave
  const int m    = lane & 15;
  const int token0 = (blockIdx.x * (int)(blockDim.x >> 5) + wave) * 16;
  const bool active = token0 < ntok; // wave-uniform; no early return (barriers below)

  const float* hrow = H + (size_t)(token0 + m) * EMB_DIM;
  const float* urow = U + (size_t)(token0 + m) * USER_DIM;
  const int swp = (m >> 3) & 1;      // B half-swap select (bank-conflict-free b128s)

  v8f acc[4];
  v8f zero = {0.f, 0.f, 0.f, 0.f, 0.f, 0.f, 0.f, 0.f};
#pragma unroll
  for (int t = 0; t < 4; ++t) acc[t] = zero;

  // One K-step of 32: 4x global_load_b128 of A (f32) per lane, cvt->bf16,
  // 2x ds_load_b128 per 16-expert tile for B, 4 WMMAs sharing the A fragment.
  // A layout: lanes 0-15 hold K 0-15, lanes 16-31 hold K 16-31.
  auto kstep = [&](const float* rowptr, int kloc, int klocal) {
    const float4* ap = (const float4*)(rowptr + kloc + half * 16);
    __builtin_prefetch(rowptr + kloc + 128, 0, 3);   // A stream, ~4 K-steps ahead
    float4 a0 = ap[0], a1 = ap[1], a2 = ap[2], a3 = ap[3];
    v16bf Av;
    Av[0]  = (__bf16)a0.x; Av[1]  = (__bf16)a0.y; Av[2]  = (__bf16)a0.z; Av[3]  = (__bf16)a0.w;
    Av[4]  = (__bf16)a1.x; Av[5]  = (__bf16)a1.y; Av[6]  = (__bf16)a1.z; Av[7]  = (__bf16)a1.w;
    Av[8]  = (__bf16)a2.x; Av[9]  = (__bf16)a2.y; Av[10] = (__bf16)a2.z; Av[11] = (__bf16)a2.w;
    Av[12] = (__bf16)a3.x; Av[13] = (__bf16)a3.y; Av[14] = (__bf16)a3.z; Av[15] = (__bf16)a3.w;
    int kbeff = (klocal >> 4) + half;  // lanes0-15 read K 0-15, lanes16-31 K 16-31
#pragma unroll
    for (int t = 0; t < 4; ++t) {
      const uint4* slot = (const uint4*)(wl + (size_t)(kbeff * 64 + 16 * t + m) * 32);
      union { uint4 q[2]; v16bf v; } Bf;
      Bf.q[0] = slot[swp];
      Bf.q[1] = slot[swp ^ 1];
      acc[t] = __builtin_amdgcn_wmma_f32_16x16x32_bf16(
          false, Av, false, Bf.v, (short)0, acc[t], false, false);
    }
  };

  // Two K-chunks: stage 1152x64 of W as bf16 into LDS, sync, run WMMAs, repeat.
  // slot(kb, n) = 32B of W[kbase+16kb+e][n], e=0..15; 16B halves swapped on (n&8)
  // so the 16 lanes of a half-wave hit all 64 LDS banks.
#pragma unroll
  for (int c = 0; c < 2; ++c) {
    if (c) __syncthreads();                 // previous chunk fully consumed
    const int kbase = c * KCHUNK;
    for (int s = threadIdx.x; s < CHUNK_SLOTS; s += 256) {
      int kb = s >> 6, n = s & 63;
      const float* wp = W + (size_t)(kbase + kb * 16) * NUM_EXPERTS + n; // coalesced in n
      unsigned p[8];
#pragma unroll
      for (int e = 0; e < 8; ++e)
        p[e] = pk_bf16(wp[(2 * e) * NUM_EXPERTS], wp[(2 * e + 1) * NUM_EXPERTS]);
      uint4 q0 = make_uint4(p[0], p[1], p[2], p[3]);   // K elems 0..7
      uint4 q1 = make_uint4(p[4], p[5], p[6], p[7]);   // K elems 8..15
      uint4* slot = (uint4*)(wl + (size_t)s * 32);
      int swap = (n >> 3) & 1;
      slot[swap]     = q0;
      slot[swap ^ 1] = q1;
    }
    __syncthreads();

    if (active) {
      const int hend = (kbase + KCHUNK < EMB_DIM) ? kbase + KCHUNK : EMB_DIM;
      for (int kg = kbase; kg < hend; kg += 32)
        kstep(hrow, kg, kg - kbase);
      const int ustart = (kbase > EMB_DIM) ? kbase : EMB_DIM;
      for (int kg = ustart; kg < kbase + KCHUNK; kg += 32)
        kstep(urow, kg - EMB_DIM, kg - kbase);
    }
  }

  if (!active) return;

  // ---- Epilogue: bias + softmax + top-8 mask + renormalize ----------------
  // C layout: v8f elem r -> lanes0-15 row r, lanes16-31 row r+8; col = 16t + m.
  float bias[4];
#pragma unroll
  for (int t = 0; t < 4; ++t) bias[t] = B[16 * t + m];

#pragma unroll
  for (int r = 0; r < 8; ++r) {
    float g[4];
#pragma unroll
    for (int t = 0; t < 4; ++t) g[t] = acc[t][r] + bias[t];

    // row max over 64 values (xor masks 1..8 stay inside each 16-lane half)
    float gm = fmaxf(fmaxf(g[0], g[1]), fmaxf(g[2], g[3]));
#pragma unroll
    for (int s = 1; s <= 8; s <<= 1) gm = fmaxf(gm, __shfl_xor(gm, s, 32));

    float e[4], Z = 0.f;
#pragma unroll
    for (int t = 0; t < 4; ++t) { e[t] = __expf(g[t] - gm); Z += e[t]; }
#pragma unroll
    for (int s = 1; s <= 8; s <<= 1) Z += __shfl_xor(Z, s, 32);

    // top-8 via 8 rounds of (value, index) arg-max; ties -> lower expert idx
    int sel = 0;
    float S = 0.f;
#pragma unroll
    for (int it = 0; it < TOPK; ++it) {
      float bv = -1.f; int bi = 0x7fffffff;
#pragma unroll
      for (int t = 0; t < 4; ++t)
        if (!(sel & (1 << t)) && e[t] > bv) { bv = e[t]; bi = 16 * t + m; }
#pragma unroll
      for (int s = 1; s <= 8; s <<= 1) {
        float ov = __shfl_xor(bv, s, 32);
        int   oi = __shfl_xor(bi, s, 32);
        if (ov > bv || (ov == bv && oi < bi)) { bv = ov; bi = oi; }
      }
      S += bv;                              // identical on all lanes of the half
      if ((bi & 15) == m) sel |= 1 << (bi >> 4);
    }

    // w_i = e_i/Z; masked renorm: (e_i/Z) / (S/Z + 1e-9) = e_i / (S + 1e-9*Z)
    float inv = 1.0f / (S + 1e-9f * Z);
    float* orow = out + (size_t)(token0 + r + 8 * half) * NUM_EXPERTS;
#pragma unroll
    for (int t = 0; t < 4; ++t)
      orow[16 * t + m] = (sel & (1 << t)) ? e[t] * inv : 0.0f;
  }
}

extern "C" void kernel_launch(void* const* d_in, const int* in_sizes, int n_in,
                              void* d_out, int out_size, void* d_ws, size_t ws_size,
                              hipStream_t stream) {
  const float* H = (const float*)d_in[0];
  const float* U = (const float*)d_in[1];
  const float* W = (const float*)d_in[2];
  const float* B = (const float*)d_in[3];
  float* out = (float*)d_out;

  int ntok   = in_sizes[0] / EMB_DIM;       // 16384
  int tiles  = (ntok + 15) / 16;            // 16-token tiles, one per wave
  int blocks = (tiles + 7) / 8;             // 8 waves (256 threads) per block

  hipFuncSetAttribute((const void*)UserAwareGate_kernel,
                      hipFuncAttributeMaxDynamicSharedMemorySize, LDS_BYTES);
  UserAwareGate_kernel<<<blocks, 256, LDS_BYTES, stream>>>(H, U, W, B, out, ntok);
}
